// AdaptiveSoftmax_25168508355076
// MI455X (gfx1250) — matrix operand order repaired and checked by reading, not compiled, not run on hardware
//
#include <hip/hip_runtime.h>
#include <hip/hip_bf16.h>

// ---------------------------------------------------------------------------
// Adaptive softmax forward on gfx1250 (MI455X).
// Stage 1: one elementwise pass converts x + all weights f32 -> bf16 (ws).
// Stage 2: five C = A * W^T GEMMs via v_wmma_f32_16x16x32_bf16 (f32 accum),
//          bf16 operands loaded directly global -> LDS (no in-loop cvt VALU),
//          double-buffered LDS, one barrier per K-tile. Epilogue variants
//          (bias / row-mask / bf16-out) are compile-time template params so
//          the stores are straight-line code.
// ---------------------------------------------------------------------------

typedef __attribute__((ext_vector_type(4)))  float          v4f;
typedef __attribute__((ext_vector_type(8)))  float          v8f;
typedef __attribute__((ext_vector_type(8)))  unsigned short v8us;
typedef __attribute__((ext_vector_type(16))) __bf16         v16bf;

#define BM 128
#define BN 128
#define BK 32
#define LDSS 40   // halves per LDS row: 32 data + 8 pad (80B, keeps 16B align)

__device__ __forceinline__ unsigned short f32_to_bf16_rne(float f) {
    unsigned int u = __builtin_bit_cast(unsigned int, f);
    u += 0x7FFFu + ((u >> 16) & 1u);
    return (unsigned short)(u >> 16);
}

union FragU {
    v8us  u[2];
    v16bf v;
};

// C[M,N] = A[M,K] * B[N,K]^T with bf16 A,B (raw ushort bits).
// OUT_BF16 ? store bf16 to Cb : store f32 to Cf.
// Requires: M % 128 == 0, K % 32 == 0. N arbitrary (guarded).
template <bool HAS_BIAS, bool HAS_SCALE, bool OUT_BF16>
__global__ __launch_bounds__(256)
void adasoft_gemm(const unsigned short* __restrict__ A,
                  const unsigned short* __restrict__ B,
                  float* __restrict__ Cf,
                  unsigned short* __restrict__ Cb,
                  int M, int N, int K,
                  const float* __restrict__ bias,
                  const float* __restrict__ rowScale)
{
    __shared__ alignas(16) unsigned short As[2][BM * LDSS];
    __shared__ alignas(16) unsigned short Bs[2][BN * LDSS];

    const int tid   = threadIdx.x;
    const int lane  = tid & 31;
    const int lhalf = lane & 15;   // column / row-in-group selector
    const int lhi   = lane >> 4;   // half-wave selector
    const int w     = tid >> 5;    // wave id 0..7
    const int wm    = (w & 3) * 32;   // wave M offset inside tile
    const int wn    = (w >> 2) * 64;  // wave N offset inside tile

    const int blockM = blockIdx.y * BM;
    const int blockN = blockIdx.x * BN;

    // Global->LDS fill: 4 lanes cover 32 halves (v8us = 16B each), 64 rows,
    // each thread handles rows gr and gr+64.
    const int cg = (tid & 3) * 8;  // k offset in halves: 0,8,16,24
    const int gr = tid >> 2;       // 0..63

    v8us a0, a1, b0, b1;

    auto loadTile = [&](int kt) {
        const int k0 = kt * BK;
        a0 = *(const v8us*)(A + (size_t)(blockM + gr) * K + k0 + cg);
        a1 = *(const v8us*)(A + (size_t)(blockM + gr + 64) * K + k0 + cg);
        const int br0 = blockN + gr, br1 = blockN + gr + 64;
        v8us z = {0, 0, 0, 0, 0, 0, 0, 0};
        b0 = (br0 < N) ? *(const v8us*)(B + (size_t)br0 * K + k0 + cg) : z;
        b1 = (br1 < N) ? *(const v8us*)(B + (size_t)br1 * K + k0 + cg) : z;
    };

    auto storeTile = [&](int buf) {
        *(v8us*)&As[buf][gr * LDSS + cg]        = a0;
        *(v8us*)&As[buf][(gr + 64) * LDSS + cg] = a1;
        *(v8us*)&Bs[buf][gr * LDSS + cg]        = b0;
        *(v8us*)&Bs[buf][(gr + 64) * LDSS + cg] = b1;
    };

    v8f acc[2][4];
#pragma unroll
    for (int i = 0; i < 2; ++i)
#pragma unroll
        for (int j = 0; j < 4; ++j) {
            v8f z = {0.f, 0.f, 0.f, 0.f, 0.f, 0.f, 0.f, 0.f};
            acc[i][j] = z;
        }

    const int ktiles = K / BK;

    // prologue: fill buffer 0
    loadTile(0);
    storeTile(0);
    __syncthreads();

    for (int kt = 0; kt < ktiles; ++kt) {
        const int  cur  = kt & 1;
        const bool more = (kt + 1 < ktiles);

        // issue next tile's global loads; they overlap with this tile's WMMAs
        if (more) loadTile(kt + 1);

        // ---- fragments per ISA 16-bit A 16x32 / B 32x16 layouts ----
        FragU af[2], bq[4];
#pragma unroll
        for (int mi = 0; mi < 2; ++mi) {
            const int ar = wm + mi * 16 + lhalf;
            af[mi].u[0] = *(const v8us*)&As[cur][ar * LDSS + 8 * lhi];
            af[mi].u[1] = *(const v8us*)&As[cur][ar * LDSS + 16 + 8 * lhi];
        }
#pragma unroll
        for (int ni = 0; ni < 4; ++ni) {
            const int br = wn + ni * 16 + lhalf;
            bq[ni].u[0] = *(const v8us*)&Bs[cur][br * LDSS + 8 * lhi];
            bq[ni].u[1] = *(const v8us*)&Bs[cur][br * LDSS + 16 + 8 * lhi];
        }

#pragma unroll
        for (int mi = 0; mi < 2; ++mi)
#pragma unroll
            for (int ni = 0; ni < 4; ++ni)
                acc[mi][ni] = __builtin_amdgcn_wmma_f32_16x16x32_bf16(
                    /*neg_a=*/false, af[mi].v,
                    /*neg_b=*/false, bq[ni].v,
                    /*c_mod=*/(short)0, acc[mi][ni],
                    /*reuse_a=*/false, /*reuse_b=*/false);

        // stores target the opposite buffer: no WAR with this tile's compute,
        // so a single barrier per K-tile suffices.
        if (more) {
            storeTile(cur ^ 1);
            __syncthreads();
        }
    }

    // ---- epilogue: straight-line, specialized at compile time ----
#pragma unroll
    for (int mi = 0; mi < 2; ++mi) {
        const int rowBase = blockM + wm + mi * 16 + lhi * 8;
        float scale[8];
        if (HAS_SCALE) {
#pragma unroll
            for (int r = 0; r < 8; ++r) scale[r] = rowScale[rowBase + r];
        }
#pragma unroll
        for (int ni = 0; ni < 4; ++ni) {
            const int col = blockN + wn + ni * 16 + lhalf;
            if (col < N) {
                float bv = 0.0f;
                if (HAS_BIAS) bv = bias[col];
#pragma unroll
                for (int r = 0; r < 8; ++r) {
                    const int row = rowBase + r;
                    float v = acc[mi][ni][r] + bv;
                    if (HAS_SCALE) v *= scale[r];
                    const size_t idx = (size_t)row * N + col;
                    if (OUT_BF16) Cb[idx] = f32_to_bf16_rne(v);
                    else          Cf[idx] = v;
                }
            }
        }
    }
}

// elementwise f32 -> bf16, n must be a multiple of 8 (true for all operands)
__global__ __launch_bounds__(256)
void adasoft_cvt_bf16(const float* __restrict__ in,
                      unsigned short* __restrict__ out, long n)
{
    long i = ((long)blockIdx.x * 256 + threadIdx.x) * 8;
    if (i < n) {
        v4f x0 = *(const v4f*)(in + i);
        v4f x1 = *(const v4f*)(in + i + 4);
        v8us h;
#pragma unroll
        for (int e = 0; e < 4; ++e) h[e]     = f32_to_bf16_rne(x0[e]);
#pragma unroll
        for (int e = 0; e < 4; ++e) h[e + 4] = f32_to_bf16_rne(x1[e]);
        *(v8us*)(out + i) = h;
    }
}

__global__ void adasoft_make_masks(const int* __restrict__ target,
                                   float* __restrict__ m0,
                                   float* __restrict__ m1, int n)
{
    int i = blockIdx.x * 256 + threadIdx.x;
    if (i < n) {
        int t = target[i];
        m0[i] = (t >= 2000 && t < 10000) ? 1.0f : 0.0f;
        m1[i] = (t >= 10000 && t < 30000) ? 1.0f : 0.0f;
    }
}

extern "C" void kernel_launch(void* const* d_in, const int* in_sizes, int n_in,
                              void* d_out, int out_size, void* d_ws, size_t ws_size,
                              hipStream_t stream)
{
    (void)in_sizes; (void)n_in; (void)out_size; (void)ws_size;

    const float* x        = (const float*)d_in[0];
    const int*   target   = (const int*)  d_in[1];
    const float* head_w   = (const float*)d_in[2];
    const float* head_b   = (const float*)d_in[3];
    const float* tail0_w1 = (const float*)d_in[4];
    const float* tail0_w2 = (const float*)d_in[5];
    const float* tail1_w1 = (const float*)d_in[6];
    const float* tail1_w2 = (const float*)d_in[7];
    float* out = (float*)d_out;

    const int NR = 8192, D = 1024;
    const int HEAD_OUT = 2002, T0 = 8000, T1 = 20000, DQ = 256;

    // ---- workspace layout (~69 MB), all 16B-aligned partitions ----
    float* mask0 = (float*)d_ws;                       //  8192 f32
    float* mask1 = mask0 + NR;                         //  8192 f32
    unsigned short* xb    = (unsigned short*)(mask1 + NR);
    unsigned short* hwb   = xb    + (size_t)NR * D;            // 8192*1024
    unsigned short* w01b  = hwb   + (size_t)HEAD_OUT * D;      // 2002*1024
    unsigned short* w02b  = w01b  + (size_t)D * D;             // 1024*1024
    unsigned short* w11b  = w02b  + (size_t)T0 * D;            // 8000*1024
    unsigned short* w12b  = w11b  + (size_t)DQ * D;            // 256*1024
    unsigned short* h0b   = w12b  + (size_t)T1 * DQ;           // 20000*256
    unsigned short* h1b   = h0b   + (size_t)NR * D;            // 8192*1024

    adasoft_make_masks<<<(NR + 255) / 256, 256, 0, stream>>>(target, mask0, mask1, NR);

    auto cvt = [&](const float* src, unsigned short* dst, long n) {
        adasoft_cvt_bf16<<<(unsigned)((n / 8 + 255) / 256), 256, 0, stream>>>(src, dst, n);
    };
    cvt(x,        xb,   (long)NR * D);
    cvt(head_w,   hwb,  (long)HEAD_OUT * D);
    cvt(tail0_w1, w01b, (long)D * D);
    cvt(tail0_w2, w02b, (long)T0 * D);
    cvt(tail1_w1, w11b, (long)DQ * D);
    cvt(tail1_w2, w12b, (long)T1 * DQ);

    const dim3 blk(256);
    const int mt = NR / BM;  // 64

    // head = x @ head_w^T + head_b            [8192, 2002] f32
    adasoft_gemm<true, false, false><<<dim3((HEAD_OUT + BN - 1) / BN, mt), blk, 0, stream>>>(
        xb, hwb, out, nullptr, NR, HEAD_OUT, D, head_b, nullptr);

    // h0 = x @ tail0_w1^T                     [8192, 1024] bf16
    adasoft_gemm<false, false, true><<<dim3(D / BN, mt), blk, 0, stream>>>(
        xb, w01b, nullptr, h0b, NR, D, D, nullptr, nullptr);

    // out0 = (h0 @ tail0_w2^T) * mask0        [8192, 8000] f32
    adasoft_gemm<false, true, false><<<dim3((T0 + BN - 1) / BN, mt), blk, 0, stream>>>(
        h0b, w02b, out + (size_t)NR * HEAD_OUT, nullptr, NR, T0, D, nullptr, mask0);

    // h1 = x @ tail1_w1^T                     [8192, 256] bf16
    adasoft_gemm<false, false, true><<<dim3(DQ / BN, mt), blk, 0, stream>>>(
        xb, w11b, nullptr, h1b, NR, DQ, D, nullptr, nullptr);

    // out1 = (h1 @ tail1_w2^T) * mask1        [8192, 20000] f32
    adasoft_gemm<false, true, false><<<dim3((T1 + BN - 1) / BN, mt), blk, 0, stream>>>(
        h1b, w12b, out + (size_t)NR * (HEAD_OUT + T0), nullptr, NR, T1, DQ, nullptr, mask1);
}
